// MultiHeadAttention_39376260170194
// MI455X (gfx1250) — matrix operand order repaired and checked by reading, not compiled, not run on hardware
//
#include <hip/hip_runtime.h>
#include <math.h>

#define B_ 4
#define H_ 16
#define S_ 2048
#define DM_ 1024
#define DK_ 64

typedef unsigned short u16;
typedef __attribute__((ext_vector_type(16))) __bf16 v16bf;
typedef __attribute__((ext_vector_type(8)))  float  v8f;

union Frag {
  v16bf v;
  u16   s[16];
  int4  q[2];
};

__device__ __forceinline__ u16 f2bf(float f) {
  unsigned u = __float_as_uint(f);
  return (u16)((u + 0x7FFFu + ((u >> 16) & 1u)) >> 16);  // RNE f32->bf16
}

__device__ __forceinline__ v8f wmma_bf16(v16bf a, v16bf b, v8f c) {
  // (neg_a, A, neg_b, B, c_mod, C, reuse_a, reuse_b)
  return __builtin_amdgcn_wmma_f32_16x16x32_bf16(false, a, false, b, (short)0, c,
                                                 false, false);
}

// A-matrix 16x32 bf16 fragment from row-major [row, k] storage (ld elements).
// ISA layout: lane = half*16+m holds K in [8*half,8*half+8) and [16+8*half, ...).
__device__ __forceinline__ v16bf load_frag_a(const u16* base, int row0, int ld, int k0) {
  int lane = threadIdx.x & 31;
  int m = lane & 15, half = lane >> 4;
  const u16* p = base + (size_t)(row0 + m) * ld + k0;
  Frag f;
  f.q[0] = *(const int4*)(p + 8 * half);
  f.q[1] = *(const int4*)(p + 16 + 8 * half);
  return f.v;
}

// B-matrix 32x16 bf16 fragment where B(k,n) = base[(col0+n)*ld + k]
// (i.e. "B stored transposed, row n contiguous over k").
// ISA layout: lane = half*16+n holds K in [8*half,8*half+8) and [16+8*half, ...).
__device__ __forceinline__ v16bf load_frag_b(const u16* base, int col0, int ld, int k0) {
  int lane = threadIdx.x & 31;
  int n = lane & 15, half = lane >> 4;
  const u16* p = base + (size_t)(col0 + n) * ld + k0;
  Frag f;
  f.q[0] = *(const int4*)(p + 8 * half);
  f.q[1] = *(const int4*)(p + 16 + 8 * half);
  return f.v;
}

// A fragment built on the fly from an f32 matrix (attn weights in d_out).
__device__ __forceinline__ v16bf load_frag_a_f32(const float* base, int row0, int ld, int k0) {
  int lane = threadIdx.x & 31;
  int m = lane & 15, half = lane >> 4;
  const float* p = base + (size_t)(row0 + m) * ld + k0;
  Frag f;
#pragma unroll
  for (int i = 0; i < 8; ++i) {
    f.s[i]     = f2bf(p[8 * half + i]);
    f.s[8 + i] = f2bf(p[16 + 8 * half + i]);
  }
  return f.v;
}

// ---------------- Pass 0: f32 -> bf16 conversion ----------------
__global__ void k_f32_to_bf16(const float* __restrict__ s, u16* __restrict__ d, int n) {
  int i = blockIdx.x * blockDim.x + threadIdx.x;
  if (i < n) d[i] = f2bf(s[i]);
}

// ---------------- Pass 1: QKV projections ----------------
// Q,K: [b,h,s,64] bf16 ; V stored transposed: [b,h,64,s] bf16.
// One wave computes a 16x64 tile; 3 mats * 16 heads * 512 row-tiles = 24576 waves.
__global__ void k_qkv(const u16* __restrict__ xb,
                      const u16* __restrict__ wqb, const u16* __restrict__ wkb,
                      const u16* __restrict__ wvb,
                      u16* __restrict__ Qb, u16* __restrict__ Kb, u16* __restrict__ Vtb) {
  int task = blockIdx.x * 8 + (threadIdx.x >> 5);
  const int RT = (B_ * S_) / 16;  // 512
  int mat = task / (H_ * RT);
  int rem = task % (H_ * RT);
  int h  = rem / RT;
  int rt = rem % RT;
  int m0 = rt * 16;
  int b  = m0 / S_;
  int s0 = m0 % S_;

  const u16* w  = (mat == 0) ? wqb : (mat == 1 ? wkb : wvb);
  const u16* wh = w + (size_t)h * DK_ * DM_;  // [64][1024], row n contiguous over k

  v8f c[4] = {};
  for (int kd = 0; kd < DM_; kd += 32) {
    v16bf a = load_frag_a(xb, m0, DM_, kd);
#pragma unroll
    for (int nt = 0; nt < 4; ++nt)
      c[nt] = wmma_bf16(a, load_frag_b(wh, nt * 16, DM_, kd), c[nt]);
  }

  int lane = threadIdx.x & 31, n = lane & 15, half = lane >> 4;
  if (mat < 2) {
    u16* dst = (mat == 0 ? Qb : Kb) + ((size_t)(b * H_ + h) * S_ + s0) * DK_;
#pragma unroll
    for (int nt = 0; nt < 4; ++nt)
#pragma unroll
      for (int r = 0; r < 8; ++r)
        dst[(size_t)(8 * half + r) * DK_ + nt * 16 + n] = f2bf(c[nt][r]);
  } else {
    u16* dst = Vtb + (size_t)(b * H_ + h) * DK_ * S_;
#pragma unroll
    for (int nt = 0; nt < 4; ++nt)
#pragma unroll
      for (int r = 0; r < 8; ++r)
        dst[(size_t)(nt * 16 + n) * S_ + s0 + 8 * half + r] = f2bf(c[nt][r]);
  }
}

// ---------------- Pass 2: fused scores + mask + softmax ----------------
// One block per (b, h, 16-query tile). 8 waves fill a 16x2048 f32 score tile in
// 128KB dynamic LDS via WMMA, then the block does the softmax and writes attn once.
__global__ void k_scores_softmax(const u16* __restrict__ Qb, const u16* __restrict__ Kb,
                                 const int* __restrict__ mask, float* __restrict__ attn) {
  extern __shared__ float smem[];        // [16][2048] scores
  __shared__ float red[256];

  int tid = threadIdx.x;
  int blk = blockIdx.x;
  const int QT = S_ / 16;  // 128
  int b  = blk / (H_ * QT);
  int rem = blk % (H_ * QT);
  int h  = rem / QT;
  int qt = rem % QT;
  int q0 = qt * 16;

  const u16* qbase = Qb + (size_t)(b * H_ + h) * S_ * DK_;
  const u16* kbase = Kb + (size_t)(b * H_ + h) * S_ * DK_;
  int warp = tid >> 5;
  int lane = tid & 31, n = lane & 15, half = lane >> 4;

  // Q fragments for this query tile are reused across all 16 s-tiles of this wave.
  v16bf a0 = load_frag_a(qbase, q0, DK_, 0);
  v16bf a1 = load_frag_a(qbase, q0, DK_, 32);

  for (int st = warp * 16; st < warp * 16 + 16; ++st) {
    int s0 = st * 16;
    v8f c = {};
    c = wmma_bf16(a0, load_frag_b(kbase, s0, DK_, 0),  c);
    c = wmma_bf16(a1, load_frag_b(kbase, s0, DK_, 32), c);
    int mk = mask[b * S_ + s0 + n];
#pragma unroll
    for (int r = 0; r < 8; ++r) {
      float v = mk ? c[r] * 0.125f : -INFINITY;  // 1/sqrt(64) = 0.125
      smem[(8 * half + r) * S_ + s0 + n] = v;
    }
  }
  __syncthreads();

  for (int row = 0; row < 16; ++row) {
    const float* srow = smem + row * S_;
    float mx = -INFINITY;
#pragma unroll
    for (int j = 0; j < 8; ++j) mx = fmaxf(mx, srow[j * 256 + tid]);
    red[tid] = mx; __syncthreads();
    for (int off = 128; off > 0; off >>= 1) {
      if (tid < off) red[tid] = fmaxf(red[tid], red[tid + off]);
      __syncthreads();
    }
    mx = red[0]; __syncthreads();

    float ssum = 0.f, ev[8];
#pragma unroll
    for (int j = 0; j < 8; ++j) {
      float xv = srow[j * 256 + tid];
      float e = (mx == -INFINITY) ? 0.f : __expf(xv - mx);  // masked row -> 0
      ev[j] = e; ssum += e;
    }
    red[tid] = ssum; __syncthreads();
    for (int off = 128; off > 0; off >>= 1) {
      if (tid < off) red[tid] += red[tid + off];
      __syncthreads();
    }
    float inv = red[0] > 0.f ? 1.f / red[0] : 0.f;
    __syncthreads();

    float* arow = attn + ((size_t)((b * H_ + h) * S_ + q0 + row)) * S_;
#pragma unroll
    for (int j = 0; j < 8; ++j) arow[j * 256 + tid] = ev[j] * inv;
  }
}

// ---------------- Pass 3: head_out = attn @ V ----------------
// One wave per 16x64 tile; A built from f32 attn on the fly, B from V^T.
// Output written directly in concat layout [b, s, h*64] (bf16).
__global__ void k_attn_v(const float* __restrict__ attn, const u16* __restrict__ Vtb,
                         u16* __restrict__ ho) {
  int task = blockIdx.x * 8 + (threadIdx.x >> 5);  // 8192
  const int QT = S_ / 16;
  int b  = task / (H_ * QT);
  int rem = task % (H_ * QT);
  int h  = rem / QT;
  int qt = rem % QT;
  int q0 = qt * 16;

  const float* abase = attn + (size_t)(b * H_ + h) * S_ * S_;
  const u16*   vt    = Vtb  + (size_t)(b * H_ + h) * DK_ * S_;

  v8f c[4] = {};
  for (int ks = 0; ks < S_; ks += 32) {
    if (ks + 128 < S_)
      __builtin_prefetch((const void*)(abase + (size_t)(q0 + (threadIdx.x & 15)) * S_ +
                                       ks + 128), 0, 0);
    v16bf a = load_frag_a_f32(abase, q0, S_, ks);
#pragma unroll
    for (int nt = 0; nt < 4; ++nt)
      c[nt] = wmma_bf16(a, load_frag_b(vt, nt * 16, S_, ks), c[nt]);
  }

  int lane = threadIdx.x & 31, n = lane & 15, half = lane >> 4;
  u16* d = ho + ((size_t)(b * S_ + q0)) * DM_ + h * DK_;
#pragma unroll
  for (int nt = 0; nt < 4; ++nt)
#pragma unroll
    for (int r = 0; r < 8; ++r)
      d[(size_t)(8 * half + r) * DM_ + nt * 16 + n] = f2bf(c[nt][r]);
}

// ---------------- Pass 4: output projection + bias ----------------
__global__ void k_oproj(const u16* __restrict__ ho, const u16* __restrict__ wob,
                        const float* __restrict__ bo, float* __restrict__ outp) {
  int task = blockIdx.x * 8 + (threadIdx.x >> 5);  // 512 row-tiles * 16 col-tiles
  const int CT = DM_ / DK_;  // 16
  int rt = task / CT;
  int ct = task % CT;
  int m0 = rt * 16;
  int n0 = ct * DK_;

  v8f c[4] = {};
  for (int kd = 0; kd < DM_; kd += 32) {
    v16bf a = load_frag_a(ho, m0, DM_, kd);
#pragma unroll
    for (int nt = 0; nt < 4; ++nt)
      c[nt] = wmma_bf16(a, load_frag_b(wob, n0 + nt * 16, DM_, kd), c[nt]);
  }

  int lane = threadIdx.x & 31, n = lane & 15, half = lane >> 4;
  float* o = outp + (size_t)m0 * DM_ + n0;
#pragma unroll
  for (int nt = 0; nt < 4; ++nt) {
    float bb = bo[n0 + nt * 16 + n];
#pragma unroll
    for (int r = 0; r < 8; ++r)
      o[(size_t)(8 * half + r) * DM_ + nt * 16 + n] = c[nt][r] + bb;
  }
}

extern "C" void kernel_launch(void* const* d_in, const int* in_sizes, int n_in,
                              void* d_out, int out_size, void* d_ws, size_t ws_size,
                              hipStream_t stream) {
  (void)in_sizes; (void)n_in; (void)out_size; (void)ws_size;
  const float* x    = (const float*)d_in[0];
  const int*   mask = (const int*)d_in[1];
  const float* wq   = (const float*)d_in[2];
  const float* wk   = (const float*)d_in[3];
  const float* wv   = (const float*)d_in[4];
  const float* wo   = (const float*)d_in[5];
  const float* bo   = (const float*)d_in[6];

  float* outp = (float*)d_out;
  float* attn = outp + (size_t)B_ * S_ * DM_;  // tuple order: (output, attn)

  // Workspace layout (bf16 as u16), total ~92 MB.
  u16* p   = (u16*)d_ws;
  u16* xb  = p; p += (size_t)B_ * S_ * DM_;
  u16* wqb = p; p += (size_t)H_ * DK_ * DM_;
  u16* wkb = p; p += (size_t)H_ * DK_ * DM_;
  u16* wvb = p; p += (size_t)H_ * DK_ * DM_;
  u16* wob = p; p += (size_t)DM_ * DM_;
  u16* Qb  = p; p += (size_t)B_ * H_ * S_ * DK_;
  u16* Kb  = p; p += (size_t)B_ * H_ * S_ * DK_;
  u16* Vtb = p; p += (size_t)B_ * H_ * DK_ * S_;
  u16* ho  = p; p += (size_t)B_ * S_ * DM_;

  // Pass 0: conversions.
  {
    int n;
    n = B_ * S_ * DM_;   k_f32_to_bf16<<<(n + 255) / 256, 256, 0, stream>>>(x,  xb,  n);
    n = H_ * DK_ * DM_;  k_f32_to_bf16<<<(n + 255) / 256, 256, 0, stream>>>(wq, wqb, n);
    n = H_ * DK_ * DM_;  k_f32_to_bf16<<<(n + 255) / 256, 256, 0, stream>>>(wk, wkb, n);
    n = H_ * DK_ * DM_;  k_f32_to_bf16<<<(n + 255) / 256, 256, 0, stream>>>(wv, wvb, n);
    n = DM_ * DM_;       k_f32_to_bf16<<<(n + 255) / 256, 256, 0, stream>>>(wo, wob, n);
  }

  // Pass 1: QKV projections (24576 wave-tasks / 8 waves per block).
  k_qkv<<<3072, 256, 0, stream>>>(xb, wqb, wkb, wvb, Qb, Kb, Vtb);

  // Pass 2: fused scores + mask + softmax (128 KB dynamic LDS per block).
  k_scores_softmax<<<B_ * H_ * (S_ / 16), 256, 16 * S_ * sizeof(float), stream>>>(
      Qb, Kb, mask, attn);

  // Pass 3: attn @ V (8192 wave-tasks).
  k_attn_v<<<1024, 256, 0, stream>>>(attn, Vtb, ho);

  // Pass 4: output projection + bias.
  k_oproj<<<1024, 256, 0, stream>>>(ho, wob, bo, outp);
}
